// NetworkTrafficModel_2731599200605
// MI455X (gfx1250) — matrix operand order repaired and checked by reading, not compiled, not run on hardware
//
#include <hip/hip_runtime.h>
#include <hip/hip_bf16.h>

typedef _Float16 half_t;
typedef __attribute__((ext_vector_type(16))) _Float16 v16h;
typedef __attribute__((ext_vector_type(8)))  float    v8f;
typedef __attribute__((ext_vector_type(4)))  float    f4v;
typedef __attribute__((ext_vector_type(4)))  _Float16 h4v;
typedef __attribute__((ext_vector_type(8)))  _Float16 h8v;
typedef __attribute__((ext_vector_type(4)))  unsigned int u32x4;
typedef __attribute__((ext_vector_type(8)))  int i32x8;
typedef __attribute__((ext_vector_type(4)))  int i32x4;

#define NN    10240          // nodes
#define BB    64             // graphs
#define EE    163840         // directed edges (before self loops)
#define E2T   (EE + NN)      // edges + self loops
#define SS    128            // sequence length
#define DIN   4096
#define TD_   256
#define NH_   4
#define HD_   64
#define NPG   160            // nodes per graph (contiguous batch ids)

#define BM 128
#define BN 64
#define LDH 40               // LDS row stride in halves (80B)

// ======================= device helpers =======================

__device__ __forceinline__ float apply_act(float x, int act) {
  if (act == 1) return 0.5f * x * (1.0f + erff(x * 0.7071067811865475f)); // exact gelu
  if (act == 2) return fmaxf(x, 0.0f);                                   // relu
  if (act == 3) return (x > 0.0f) ? x : expm1f(x);                       // elu
  return x;
}

// A fragment: 16x32 f16, ISA 7.12.2 (lanes 0-15: K pairs 0..7 & 16..23; lanes 16-31: +8)
__device__ __forceinline__ v16h load_frag_a(const half_t* As, int lane) {
  int row  = lane & 15;
  int kofs = (lane & 16) ? 8 : 0;
  const half_t* r = As + row * LDH + kofs;
  v16h a;
#pragma unroll
  for (int g = 0; g < 8; ++g) {
    int k = (g < 4) ? (2 * g) : (16 + 2 * (g - 4));
    a[2 * g]     = r[k];
    a[2 * g + 1] = r[k + 1];
  }
  return a;
}

// B fragment from transposed LDS tile Bt[n][k]: lanes 0-15 K=0..7 & 16..23, lanes 16-31 +8
__device__ __forceinline__ v16h load_frag_b(const half_t* Bt, int lane) {
  int n    = lane & 15;
  int kofs = (lane & 16) ? 8 : 0;
  const half_t* r = Bt + n * LDH + kofs;
  v16h b;
#pragma unroll
  for (int g = 0; g < 8; ++g) {
    int k = (g < 4) ? (2 * g) : (16 + 2 * (g - 4));
    b[2 * g]     = r[k];
    b[2 * g + 1] = r[k + 1];
  }
  return b;
}

__device__ __forceinline__ void edge_sd(const int* ei, int e, int& s, int& d) {
  if (e < EE) { s = ei[e]; d = ei[EE + e]; }
  else        { s = e - EE; d = e - EE; }        // appended self loops
}

// ======================= WMMA GEMM =======================
// C[M,N] = act(A[M,K-range] @ B[K,N] + bias[N]); A row stride lda, C stride ldc.
// (kend-kbeg) % 32 == 0. bias may be null (raw partial output for split-K).
// 256 threads = 8 waves; tile 128x64; wave computes 32x32 (4 x 16x16x32 WMMA).
__global__ __launch_bounds__(256) void gemm_kernel(
    const float* __restrict__ A, const float* __restrict__ B,
    const float* __restrict__ bias, float* __restrict__ C,
    int M, int N, int lda, int ldc, int act, int kbeg, int kend) {
  __shared__ half_t As[BM][LDH];
  __shared__ half_t Bt[BN][LDH];
  const int tid  = threadIdx.x;
  const int lane = tid & 31;
  const int wave = tid >> 5;
  const int mbase = blockIdx.y * BM;
  const int nbase = blockIdx.x * BN;
  const int wr = (wave >> 1) * 32;   // 0,32,64,96
  const int wc = (wave & 1) * 32;    // 0,32
  const bool full = (mbase + BM <= M) && (nbase + BN <= N) && ((N & 3) == 0);

  v8f acc00 = {}, acc01 = {}, acc10 = {}, acc11 = {};

  auto mm = [&]() {
    v16h a0 = load_frag_a(&As[wr][0], lane);
    v16h a1 = load_frag_a(&As[wr + 16][0], lane);
    v16h b0 = load_frag_b(&Bt[wc][0], lane);
    v16h b1 = load_frag_b(&Bt[wc + 16][0], lane);
    acc00 = __builtin_amdgcn_wmma_f32_16x16x32_f16(false, a0, false, b0, (short)0, acc00, false, false);
    acc01 = __builtin_amdgcn_wmma_f32_16x16x32_f16(false, a0, false, b1, (short)0, acc01, false, false);
    acc10 = __builtin_amdgcn_wmma_f32_16x16x32_f16(false, a1, false, b0, (short)0, acc10, false, false);
    acc11 = __builtin_amdgcn_wmma_f32_16x16x32_f16(false, a1, false, b1, (short)0, acc11, false, false);
  };

  if (full) {
    // -------- branch-free full-tile loop --------
    for (int k0 = kbeg; k0 < kend; k0 += 32) {
#pragma unroll
      for (int i = 0; i < 4; ++i) {
        int i4 = tid + i * 256;                 // 1024 float4s of the A panel
        int r = i4 >> 3, c4 = (i4 & 7) << 2;
        const float* ap = A + (long)(mbase + r) * lda + k0 + c4;
        f4v v = *(const f4v*)ap;
        if (k0 + 32 < kend) __builtin_prefetch(ap + 32, 0, 1);
        h4v hv;
        hv[0] = (half_t)v[0]; hv[1] = (half_t)v[1];
        hv[2] = (half_t)v[2]; hv[3] = (half_t)v[3];
        *(h4v*)&As[r][c4] = hv;
      }
      {
        int n = tid & 63, kq = tid >> 6;        // kq in 0..3
        const float* bp = B + (long)(k0 + kq * 8) * N + nbase + n;
        h8v hv;
#pragma unroll
        for (int j = 0; j < 8; ++j) { hv[j] = (half_t)(*bp); bp += N; }
        *(h8v*)&Bt[n][kq * 8] = hv;
      }
      __syncthreads();
      mm();
      __syncthreads();
    }
  } else {
    // -------- guarded edge-tile loop --------
    for (int k0 = kbeg; k0 < kend; k0 += 32) {
#pragma unroll
      for (int i = 0; i < 16; ++i) {
        int id = tid + i * 256;
        int r = id >> 5, c = id & 31;
        int row = mbase + r;
        float v = (row < M) ? A[(long)row * lda + k0 + c] : 0.0f;
        As[r][c] = (half_t)v;
      }
#pragma unroll
      for (int i = 0; i < 8; ++i) {
        int id = tid + i * 256;
        int kk = id >> 6, nn = id & 63;
        int col = nbase + nn;
        float v = (col < N) ? B[(long)(k0 + kk) * N + col] : 0.0f;
        Bt[nn][kk] = (half_t)v;
      }
      __syncthreads();
      mm();
      __syncthreads();
    }
  }

  // C/D layout: VGPR g -> M = g + (lane>=16 ? 8 : 0), N = lane & 15
  const int nloc = lane & 15;
  const int mofs = (lane & 16) ? 8 : 0;
  const int c0 = nbase + wc + nloc;
  const int c1 = c0 + 16;
  if (full) {
    float bv0 = bias ? bias[c0] : 0.0f;
    float bv1 = bias ? bias[c1] : 0.0f;
#pragma unroll
    for (int g = 0; g < 8; ++g) {
      int r0 = mbase + wr + mofs + g;
      int r1 = r0 + 16;
      C[(long)r0 * ldc + c0] = apply_act(acc00[g] + bv0, act);
      C[(long)r0 * ldc + c1] = apply_act(acc01[g] + bv1, act);
      C[(long)r1 * ldc + c0] = apply_act(acc10[g] + bv0, act);
      C[(long)r1 * ldc + c1] = apply_act(acc11[g] + bv1, act);
    }
  } else {
    float bv0 = (bias && c0 < N) ? bias[c0] : 0.0f;
    float bv1 = (bias && c1 < N) ? bias[c1] : 0.0f;
#pragma unroll
    for (int g = 0; g < 8; ++g) {
      int r0 = mbase + wr + mofs + g;
      int r1 = r0 + 16;
      if (r0 < M && c0 < N) C[(long)r0 * ldc + c0] = apply_act(acc00[g] + bv0, act);
      if (r0 < M && c1 < N) C[(long)r0 * ldc + c1] = apply_act(acc01[g] + bv1, act);
      if (r1 < M && c0 < N) C[(long)r1 * ldc + c0] = apply_act(acc10[g] + bv0, act);
      if (r1 < M && c1 < N) C[(long)r1 * ldc + c1] = apply_act(acc11[g] + bv1, act);
    }
  }
}

// sum split-K partials + bias + activation
__global__ void splitk_reduce_kernel(const float* __restrict__ part, const float* __restrict__ bias,
                                     float* __restrict__ out, long total, int N, int nsplit, int act) {
  long idx = (long)blockIdx.x * blockDim.x + threadIdx.x;
  if (idx >= total) return;
  float s = 0.0f;
  for (int p = 0; p < nsplit; ++p) s += part[(long)p * total + idx];
  out[idx] = apply_act(s + bias[(int)(idx % N)], act);
}

// ======================= small utility kernels =======================

__global__ void fill_int_kernel(int* p, int v, long n) {
  long i = (long)blockIdx.x * blockDim.x + threadIdx.x;
  if (i < n) p[i] = v;
}

__global__ void copy_cols_kernel(const float* src, float* dst, int rows, int cols,
                                 int lds, int ldd) {
  long i = (long)blockIdx.x * blockDim.x + threadIdx.x;
  if (i >= (long)rows * cols) return;
  int r = (int)(i / cols), c = (int)(i % cols);
  dst[(long)r * ldd + c] = src[(long)r * lds + c];
}

// ======================= CSR build (shared by all 4 GAT layers) =======================

__global__ void hist_kernel(const int* __restrict__ ei, int* __restrict__ deg) {
  int e = blockIdx.x * blockDim.x + threadIdx.x;
  if (e >= E2T) return;
  int s, d; edge_sd(ei, e, s, d);
  atomicAdd(&deg[d], 1);
}

// single-block exclusive scan over NN entries (40 chunks of 256)
__global__ __launch_bounds__(256) void scan_kernel(const int* __restrict__ deg,
                                                   int* __restrict__ basev,
                                                   int* __restrict__ cursor) {
  __shared__ int buf[256];
  __shared__ int carry_s;
  int t = threadIdx.x;
  if (t == 0) carry_s = 0;
  __syncthreads();
  for (int chunk = 0; chunk < NN; chunk += 256) {
    int v = deg[chunk + t];
    buf[t] = v;
    __syncthreads();
    for (int o = 1; o < 256; o <<= 1) {
      int x = (t >= o) ? buf[t - o] : 0;
      __syncthreads();
      buf[t] += x;
      __syncthreads();
    }
    int excl = carry_s + buf[t] - v;
    basev[chunk + t]  = excl;
    cursor[chunk + t] = excl;
    int total = buf[255];
    __syncthreads();
    if (t == 0) carry_s += total;
    __syncthreads();
  }
  if (t == 0) basev[NN] = E2T;
}

__global__ void scatter_kernel(const int* __restrict__ ei, int* __restrict__ cursor,
                               int* __restrict__ eids) {
  int e = blockIdx.x * blockDim.x + threadIdx.x;
  if (e >= E2T) return;
  int s, d; edge_sd(ei, e, s, d);
  int p = atomicAdd(&cursor[d], 1);
  eids[p] = e;
}

// ======================= GATv2 passes (CSR, no float atomics) =======================

__global__ void gat_logits_kernel(const float* __restrict__ xl, const float* __restrict__ xr,
                                  const float* __restrict__ att, const int* __restrict__ ei,
                                  int H, int Cc, float* __restrict__ logits) {
  long idx = (long)blockIdx.x * blockDim.x + threadIdx.x;
  if (idx >= (long)E2T * H) return;
  int e = (int)(idx / H), h = (int)(idx % H);
  int s, d; edge_sd(ei, e, s, d);
  const float* pl = xl + (long)s * H * Cc + h * Cc;
  const float* pr = xr + (long)d * H * Cc + h * Cc;
  const float* pa = att + h * Cc;
  float acc = 0.0f;
  for (int c = 0; c < Cc; ++c) {
    float m = pl[c] + pr[c];
    m = (m > 0.0f) ? m : 0.2f * m;          // leaky_relu 0.2
    acc += m * pa[c];
  }
  logits[idx] = acc;
}

// one thread per (node, head): softmax over in-edges, writes alpha per edge
__global__ void gat_softmax_kernel(const float* __restrict__ logits,
                                   const int* __restrict__ basev, const int* __restrict__ eids,
                                   int H, float* __restrict__ alpha) {
  long idx = (long)blockIdx.x * blockDim.x + threadIdx.x;
  if (idx >= (long)NN * H) return;
  int nid = (int)(idx / H), h = (int)(idx % H);
  int b0 = basev[nid], b1 = basev[nid + 1];
  float m = -3.0e38f;
  for (int i = b0; i < b1; ++i) m = fmaxf(m, logits[(long)eids[i] * H + h]);
  float z = 0.0f;
  for (int i = b0; i < b1; ++i) z += expf(logits[(long)eids[i] * H + h] - m);
  float invz = 1.0f / z;
  for (int i = b0; i < b1; ++i) {
    int e = eids[i];
    alpha[(long)e * H + h] = expf(logits[(long)e * H + h] - m) * invz;
  }
}

// one block (128 thr) per node: out[nid,:] = elu(sum_e alpha * xl[src] + bias)
__global__ __launch_bounds__(128) void gat_aggregate_kernel(
    const float* __restrict__ xl, const float* __restrict__ alpha,
    const int* __restrict__ basev, const int* __restrict__ eids,
    const int* __restrict__ ei, const float* __restrict__ bias,
    int H, int Cc, float* __restrict__ out, int ldc) {
  int nid = blockIdx.x;
  int t = threadIdx.x;
  int HC = H * Cc;
  int per = HC >> 7;                         // 4 (layer1) or 1 (layer2)
  float acc[4] = {0.0f, 0.0f, 0.0f, 0.0f};
  int b0 = basev[nid], b1 = basev[nid + 1];
  for (int i = b0; i < b1; ++i) {
    int e = eids[i];
    int s, d; edge_sd(ei, e, s, d);
    const float* xs = xl + (long)s * HC;
    const float* al = alpha + (long)e * H;
    for (int p = 0; p < per; ++p) {
      int j = t + (p << 7);
      acc[p] += xs[j] * al[j / Cc];
    }
  }
  for (int p = 0; p < per; ++p) {
    int j = t + (p << 7);
    float v = acc[p] + bias[j];
    out[(long)nid * ldc + j] = (v > 0.0f) ? v : expm1f(v);   // elu
  }
}

// ======================= pooling / elementwise =======================

__global__ void pool_mean_kernel(const float* __restrict__ nodes, float* __restrict__ gg) {
  int idx = blockIdx.x * blockDim.x + threadIdx.x;
  if (idx >= BB * TD_) return;
  int b = idx / TD_, j = idx % TD_;
  float s = 0.0f;
  for (int i = 0; i < NPG; ++i) s += nodes[(long)(b * NPG + i) * TD_ + j];
  gg[idx] = s * (1.0f / (float)NPG);
}

__global__ void bn_relu_kernel(float* __restrict__ h, const float* __restrict__ g,
                               const float* __restrict__ b, long n, int D) {
  long idx = (long)blockIdx.x * blockDim.x + threadIdx.x;
  if (idx >= n) return;
  int c = (int)(idx % D);
  float inv = rsqrtf(1.0f + 1e-5f);
  float v = h[idx] * inv * g[c] + b[c];
  h[idx] = fmaxf(v, 0.0f);
}

__global__ void pe_scale_kernel(float* __restrict__ x) {
  long idx = (long)blockIdx.x * blockDim.x + threadIdx.x;
  if (idx >= (long)BB * SS * TD_) return;
  int d = (int)(idx % TD_);
  int s = (int)((idx / TD_) % SS);
  int i = d >> 1;
  float div = expf(-(float)(2 * i) * (logf(10000.0f) / (float)TD_));
  float ang = (float)s * div;
  float pe = (d & 1) ? cosf(ang) : sinf(ang);
  x[idx] = x[idx] * 16.0f + pe;               // sqrt(256) = 16
}

__global__ void kpm_kernel(const float* __restrict__ mask, float* __restrict__ kpm) {
  int idx = blockIdx.x * blockDim.x + threadIdx.x;
  if (idx >= BB * SS) return;
  const float* m = mask + (long)idx * 64;
  float s = 0.0f;
  for (int j = 0; j < 64; ++j) s += m[j];
  kpm[idx] = (s == 64.0f) ? 1.0f : 0.0f;
}

// ======================= attention =======================

// One block (64 thr = 2 waves) per (b, h). K/V head slices (128x64 f32, row stride 768)
// staged into LDS by the Tensor Data Mover; Q staged cooperatively with padded stride.
// Each thread owns one query row per phase -> private softmax, no reductions.
// Dynamic LDS: Ks 8192 | Vs 8192 | Qs 128*65 | Ps 64*132 floats.
#define SA_KS   0
#define SA_VS   8192
#define SA_QS   16384
#define SA_PS   (16384 + 128 * 65)
#define SA_LDS_FLOATS (SA_PS + 64 * 132)

__global__ __launch_bounds__(64) void self_attn_kernel(
    const float* __restrict__ qkv, const float* __restrict__ kpm, float* __restrict__ out) {
  extern __shared__ float smem[];
  float* Ks = smem + SA_KS;
  float* Vs = smem + SA_VS;
  float* Qs = smem + SA_QS;
  float* Ps = smem + SA_PS;
  int b = blockIdx.x / NH_;
  int h = blockIdx.x % NH_;
  int t = threadIdx.x;
  const float* base = qkv + (long)(b * SS) * (3 * TD_) + h * HD_;

#if __has_builtin(__builtin_amdgcn_tensor_load_to_lds) && __has_builtin(__builtin_amdgcn_s_wait_tensorcnt)
  if ((t >> 5) == 0) {                    // wave 0 issues both TDM loads (EXEC ignored by TDM)
    unsigned ldsbase = (unsigned)(unsigned long long)(void*)smem;
#pragma unroll
    for (int sect = 0; sect < 2; ++sect) {        // 0 -> K slice, 1 -> V slice
      unsigned long long ga = (unsigned long long)(base + (sect + 1) * TD_);
      u32x4 g0;
      g0[0] = 1u;                                               // count=1, user mode
      g0[1] = ldsbase + (unsigned)((sect ? SA_VS : SA_KS) * 4); // lds_addr (bytes)
      g0[2] = (unsigned)(ga & 0xffffffffu);                     // global_addr[31:0]
      g0[3] = (unsigned)((ga >> 32) & 0x01ffffffu) | (2u << 30);// addr[56:32] | type=2
      i32x8 g1;
      g1[0] = 0x00020000;              // data_size=2 (4B), no multicast/pad
      g1[1] = (int)(64u << 16);        // tensor_dim0 = 64 (lo16 at bits 63:48)
      g1[2] = (int)(128u << 16);       // tensor_dim0 hi | tensor_dim1 = 128 (lo16)
      g1[3] = (int)(64u << 16);        // tensor_dim1 hi | tile_dim0 = 64
      g1[4] = 128;                     // tile_dim1 = 128, tile_dim2 = 0
      g1[5] = 3 * TD_;                 // tensor_dim0_stride = 768 (lo32)
      g1[6] = 0;
      g1[7] = 0;
      i32x4 gz  = {0, 0, 0, 0};
      i32x8 gz8 = {0, 0, 0, 0, 0, 0, 0, 0};
      __builtin_amdgcn_tensor_load_to_lds(g0, g1, gz, gz, gz8, 0);
    }
    __builtin_amdgcn_s_wait_tensorcnt(0);
  }
#else
  for (int i = t; i < SS * HD_; i += 64) {
    int r = i >> 6, c = i & 63;
    Ks[i] = base[(long)r * (3 * TD_) + TD_ + c];
    Vs[i] = base[(long)r * (3 * TD_) + 2 * TD_ + c];
  }
#endif
  // Q slice staged with stride 65 to avoid bank conflicts on per-thread row reads
  for (int i = t; i < SS * HD_; i += 64) {
    int r = i >> 6, c = i & 63;
    Qs[r * 65 + c] = base[(long)r * (3 * TD_) + c];
  }
  __syncthreads();

  for (int ph = 0; ph < 2; ++ph) {
    int q = (ph << 6) + t;
    const float* qr = Qs + q * 65;
    float* pr = Ps + t * 132;
    for (int k = 0; k < SS; ++k) {
      const float* kr = Ks + k * HD_;          // broadcast across lanes
      float sdot = 0.0f;
      for (int d = 0; d < HD_; ++d) sdot += qr[d] * kr[d];
      sdot *= 0.125f;                          // 1/sqrt(64)
      if (kpm[b * SS + k] != 0.0f) sdot = -1e9f;
      pr[k] = sdot;
    }
    float m = -3.0e38f;
    for (int k = 0; k < SS; ++k) m = fmaxf(m, pr[k]);
    float z = 0.0f;
    for (int k = 0; k < SS; ++k) { float e = expf(pr[k] - m); pr[k] = e; z += e; }
    float invz = 1.0f / z;
    float* orow = out + (long)(b * SS + q) * TD_ + h * HD_;
    for (int d = 0; d < HD_; ++d) {
      float acc = 0.0f;
      for (int k = 0; k < SS; ++k) acc += pr[k] * Vs[k * HD_ + d];  // Vs broadcast
      orow[d] = acc * invz;
    }
  }
}

// one block (128 thr) per (b, h); query length 1
__global__ __launch_bounds__(128) void cross_attn_kernel(
    const float* __restrict__ Q, const float* __restrict__ Kb, const float* __restrict__ Vb,
    float* __restrict__ out) {
  __shared__ float qv[HD_];
  __shared__ float sc[SS];
  __shared__ float red[SS];
  int b = blockIdx.x / NH_;
  int h = blockIdx.x % NH_;
  int t = threadIdx.x;

  if (t < HD_) qv[t] = Q[(long)b * TD_ + h * HD_ + t];
  __syncthreads();

  const float* krow = Kb + (long)(b * SS + t) * TD_ + h * HD_;
  float sdot = 0.0f;
  for (int d = 0; d < HD_; ++d) sdot += qv[d] * krow[d];
  sdot *= 0.125f;
  sc[t] = sdot;
  red[t] = sdot;
  __syncthreads();
  for (int o = 64; o > 0; o >>= 1) { if (t < o) red[t] = fmaxf(red[t], red[t + o]); __syncthreads(); }
  float mx = red[0];
  __syncthreads();
  float e = expf(sc[t] - mx);
  sc[t] = e;
  red[t] = e;
  __syncthreads();
  for (int o = 64; o > 0; o >>= 1) { if (t < o) red[t] += red[t + o]; __syncthreads(); }
  float z = red[0];
  __syncthreads();

  if (t < HD_) {
    float o = 0.0f;
    for (int kk = 0; kk < SS; ++kk)
      o += sc[kk] * Vb[(long)(b * SS + kk) * TD_ + h * HD_ + t];
    out[(long)b * TD_ + h * HD_ + t] = o / z;
  }
}

// block per row, blockDim = D (power of two); optional residual add + post-act
__global__ void ln_kernel(const float* __restrict__ x, const float* __restrict__ res,
                          const float* __restrict__ g, const float* __restrict__ b,
                          float* __restrict__ out, int D, int act) {
  __shared__ float red[256];
  int row = blockIdx.x, t = threadIdx.x;
  float v = x[(long)row * D + t] + (res ? res[(long)row * D + t] : 0.0f);
  red[t] = v;
  __syncthreads();
  for (int o = D >> 1; o > 0; o >>= 1) { if (t < o) red[t] += red[t + o]; __syncthreads(); }
  float mean = red[0] / (float)D;
  __syncthreads();
  float dv = v - mean;
  red[t] = dv * dv;
  __syncthreads();
  for (int o = D >> 1; o > 0; o >>= 1) { if (t < o) red[t] += red[t + o]; __syncthreads(); }
  float var = red[0] / (float)D;
  __syncthreads();
  float y = dv * rsqrtf(var + 1e-5f) * g[t] + b[t];
  out[(long)row * D + t] = apply_act(y, act);
}

// ======================= host orchestration =======================

static inline void launch_gemm(hipStream_t st, const float* A, const float* B,
                               const float* bias, float* C, int M, int N, int K,
                               int ldc, int act) {
  dim3 grid((N + BN - 1) / BN, (M + BM - 1) / BM);
  gemm_kernel<<<grid, 256, 0, st>>>(A, B, bias, C, M, N, K, ldc, act, 0, K);
}

static inline dim3 g1d(long n) { return dim3((unsigned)((n + 255) / 256)); }

static void gat_layer(hipStream_t st, const float* x, int in_dim,
                      const float* Wl, const float* bl, const float* Wr, const float* br,
                      const float* att, const float* bias, int H, int Cc,
                      const int* eidx, const int* basev, const int* eids,
                      float* xl, float* xr, float* logits, float* alpha,
                      float* out, int ldc) {
  int HC = H * Cc;
  launch_gemm(st, x, Wl, bl, xl, NN, HC, in_dim, HC, 0);
  launch_gemm(st, x, Wr, br, xr, NN, HC, in_dim, HC, 0);
  gat_logits_kernel<<<g1d((long)E2T * H), 256, 0, st>>>(xl, xr, att, eidx, H, Cc, logits);
  gat_softmax_kernel<<<g1d((long)NN * H), 256, 0, st>>>(logits, basev, eids, H, alpha);
  gat_aggregate_kernel<<<NN, 128, 0, st>>>(xl, alpha, basev, eids, eidx, bias, H, Cc, out, ldc);
}

extern "C" void kernel_launch(void* const* d_in, const int* in_sizes, int n_in,
                              void* d_out, int out_size, void* d_ws, size_t ws_size,
                              hipStream_t stream) {
  (void)in_sizes; (void)n_in; (void)out_size; (void)ws_size;

  const float* x_packet = (const float*)d_in[0];
  const float* x_time   = (const float*)d_in[1];
  const int*   eidx     = (const int*)d_in[2];
  // d_in[3] = batch (unused: graph ids are contiguous, 160 nodes each)
  const float* mainm    = (const float*)d_in[4];   // (64,128,64) -> (8192,64)
  const float* mainmask = (const float*)d_in[5];

  // params: JAX pytree leaf order (dict keys sorted alphabetically, lists in order)
  // 6..11 cls{W1,W2,b,b1,b2,g}; 12..19 cross{Wk,Wo,Wq,Wv,bk,bo,bq,bv};
  // 20..23 enh{W1,W2,b1,b2}; 24..27 fe_p{W1,W2,b1,b2}; 28..31 fe_t; 32..35 fus;
  // 36..47 gat_p[0..1]{Wl,Wr,att,bias,bl,br}; 48..59 gat_t; 60..83 layers[0..1]
  // {Wf1,Wf2,Wo,Wqkv,be1,be2,bf1,bf2,bo,bqkv,g1,g2}; 84..89 proj{W1,W2,b1,b2,bbn,gbn}
  const float* P[96];
  for (int i = 6; i < 90; ++i) P[i] = (const float*)d_in[i];
  const float *cls_W1 = P[6], *cls_W2 = P[7], *cls_b = P[8], *cls_b1 = P[9],
              *cls_b2 = P[10], *cls_g = P[11];
  const float *cr_Wk = P[12], *cr_Wo = P[13], *cr_Wq = P[14], *cr_Wv = P[15],
              *cr_bk = P[16], *cr_bo = P[17], *cr_bq = P[18], *cr_bv = P[19];
  const float *enh_W1 = P[20], *enh_W2 = P[21], *enh_b1 = P[22], *enh_b2 = P[23];
  const float *fus_W1 = P[32], *fus_W2 = P[33], *fus_b1 = P[34], *fus_b2 = P[35];
  const float *pr_W1 = P[84], *pr_W2 = P[85], *pr_b1 = P[86], *pr_b2 = P[87],
              *pr_bbn = P[88], *pr_gbn = P[89];

  // -------- workspace layout (floats/ints), scratch region overlaid --------
  size_t off = 0;
  auto alloc = [&](long elems) -> float* {
    float* p = (float*)((char*)d_ws + off);
    off += (((size_t)elems * 4 + 255) / 256) * 256;
    return p;
  };
  float* R = alloc(21708800L);           // overlaid scratch region
  // branch view into R:
  float* S_H1  = R;                      // NN*128
  float* S_X   = R + 1310720L;           // NN*64
  float* S_XL  = R + 1966080L;           // NN*512 (also split-K partials for GEMM1)
  float* S_XR  = R + 7208960L;           // NN*512
  float* S_G1  = R + 12451840L;          // NN*512
  float* S_XL2 = R + 17694720L;          // NN*128
  float* S_XR2 = R + 19005440L;          // NN*128
  float* S_LOG = R + 20316160L;          // E2T*4
  float* S_E   = R + 21012480L;          // E2T*4
  // transformer view into R (branches done by then):
  float* QKV   = R;                      // 8192*768
  float* ATTO  = R + 6291456L;           // 8192*256
  float* ATTO2 = R + 8388608L;           // 8192*256
  float* FF1   = R + 10485760L;          // 8192*1024

  float* NODES  = alloc((long)NN * 256); // concat(x_p, x_t)
  float* ENH1   = alloc((long)NN * 512);
  float* NODES2 = alloc((long)NN * 256);
  float* GG     = alloc(BB * TD_);
  float* TX     = alloc((long)BB * SS * TD_);
  float* KPM    = alloc(BB * SS);
  float* CR_K   = alloc((long)BB * SS * TD_);
  float* CR_V   = alloc((long)BB * SS * TD_);
  float* CR_Q   = alloc(BB * TD_);
  float* ATTN   = alloc(BB * TD_);
  float* ATTN2  = alloc(BB * TD_);
  float* FUSCAT = alloc(BB * 512);
  float* FUS1   = alloc(BB * 1024);
  float* FUSED  = alloc(BB * TD_);
  float* CLS1   = alloc(BB * 128);
  int* DEG    = (int*)alloc(NN);
  int* BASEV  = (int*)alloc(NN + 1);
  int* CURSOR = (int*)alloc(NN);
  int* EIDS   = (int*)alloc(E2T);

  // -------- CSR over (edges + self loops), shared by all 4 GAT layers --------
  fill_int_kernel<<<g1d(NN), 256, 0, stream>>>(DEG, 0, NN);
  hist_kernel<<<g1d(E2T), 256, 0, stream>>>(eidx, DEG);
  scan_kernel<<<1, 256, 0, stream>>>(DEG, BASEV, CURSOR);
  scatter_kernel<<<g1d(E2T), 256, 0, stream>>>(eidx, CURSOR, EIDS);

  // -------- dual GATv2 branches --------
  for (int br = 0; br < 2; ++br) {
    const float* xraw = br ? x_time : x_packet;
    int fb = br ? 28 : 24;   // fe params base
    int gb = br ? 48 : 36;   // gat params base
    // extractor GEMM1 (10240x128x4096) with split-K x4 for WGP occupancy;
    // partials land in S_XL (free until GAT layer 1), reduce fuses bias+GELU.
    {
      dim3 grid((128 + BN - 1) / BN, (NN + BM - 1) / BM);
      for (int s = 0; s < 4; ++s)
        gemm_kernel<<<grid, 256, 0, stream>>>(xraw, P[fb + 0], nullptr,
                                              S_XL + (long)s * NN * 128,
                                              NN, 128, DIN, 128, 0,
                                              s * 1024, (s + 1) * 1024);
      splitk_reduce_kernel<<<g1d((long)NN * 128), 256, 0, stream>>>(
          S_XL, P[fb + 2], S_H1, (long)NN * 128, 128, 4, 1);
    }
    launch_gemm(stream, S_H1, P[fb + 1], P[fb + 3], S_X, NN, 64, 128, 64, 0);
    // GAT layer 1: 64 -> 4x128, elu
    gat_layer(stream, S_X, 64, P[gb + 0], P[gb + 4], P[gb + 1], P[gb + 5],
              P[gb + 2], P[gb + 3], 4, 128, eidx, BASEV, EIDS,
              S_XL, S_XR, S_LOG, S_E, S_G1, 512);
    // GAT layer 2: 512 -> 1x128, elu, written into NODES concat columns
    gat_layer(stream, S_G1, 512, P[gb + 6], P[gb + 10], P[gb + 7], P[gb + 11],
              P[gb + 8], P[gb + 9], 1, 128, eidx, BASEV, EIDS,
              S_XL2, S_XR2, S_LOG, S_E, NODES + br * 128, 256);
  }

  // -------- node enhancement + mean pool --------
  launch_gemm(stream, NODES, enh_W1, enh_b1, ENH1, NN, 512, 256, 512, 1);
  launch_gemm(stream, ENH1, enh_W2, enh_b2, NODES2, NN, 256, 512, 256, 0);
  pool_mean_kernel<<<g1d(BB * TD_), 256, 0, stream>>>(NODES2, GG);

  // -------- temporal projection + PE --------
  launch_gemm(stream, mainm, pr_W1, pr_b1, ATTO, BB * SS, 256, 64, 256, 0);
  bn_relu_kernel<<<g1d((long)BB * SS * 256), 256, 0, stream>>>(ATTO, pr_gbn, pr_bbn,
                                                               (long)BB * SS * 256, 256);
  launch_gemm(stream, ATTO, pr_W2, pr_b2, TX, BB * SS, TD_, 256, TD_, 0);
  pe_scale_kernel<<<g1d((long)BB * SS * TD_), 256, 0, stream>>>(TX);
  kpm_kernel<<<g1d(BB * SS), 256, 0, stream>>>(mainmask, KPM);

  // -------- transformer encoder (2 post-norm layers) --------
  for (int li = 0; li < 2; ++li) {
    int lb = 60 + 12 * li;
    const float *Wf1 = P[lb + 0], *Wf2 = P[lb + 1], *Wo = P[lb + 2], *Wqkv = P[lb + 3];
    const float *be1 = P[lb + 4], *be2 = P[lb + 5], *bf1 = P[lb + 6], *bf2 = P[lb + 7];
    const float *bo = P[lb + 8], *bqkv = P[lb + 9], *g1 = P[lb + 10], *g2 = P[lb + 11];
    launch_gemm(stream, TX, Wqkv, bqkv, QKV, BB * SS, 3 * TD_, TD_, 3 * TD_, 0);
    self_attn_kernel<<<BB * NH_, 64, SA_LDS_FLOATS * 4, stream>>>(QKV, KPM, ATTO);
    launch_gemm(stream, ATTO, Wo, bo, ATTO2, BB * SS, TD_, TD_, TD_, 0);
    ln_kernel<<<BB * SS, TD_, 0, stream>>>(ATTO2, TX, g1, be1, TX, TD_, 0);
    launch_gemm(stream, TX, Wf1, bf1, FF1, BB * SS, 4 * TD_, TD_, 4 * TD_, 2);
    launch_gemm(stream, FF1, Wf2, bf2, ATTO2, BB * SS, TD_, 4 * TD_, TD_, 0);
    ln_kernel<<<BB * SS, TD_, 0, stream>>>(ATTO2, TX, g2, be2, TX, TD_, 0);
  }

  // -------- cross attention (query = pooled graph feature) --------
  launch_gemm(stream, GG, cr_Wq, cr_bq, CR_Q, BB, TD_, TD_, TD_, 0);
  launch_gemm(stream, TX, cr_Wk, cr_bk, CR_K, BB * SS, TD_, TD_, TD_, 0);
  launch_gemm(stream, TX, cr_Wv, cr_bv, CR_V, BB * SS, TD_, TD_, TD_, 0);
  cross_attn_kernel<<<BB * NH_, 128, 0, stream>>>(CR_Q, CR_K, CR_V, ATTN);
  launch_gemm(stream, ATTN, cr_Wo, cr_bo, ATTN2, BB, TD_, TD_, TD_, 0);

  // -------- fusion --------
  copy_cols_kernel<<<g1d(BB * TD_), 256, 0, stream>>>(ATTN2, FUSCAT, BB, TD_, TD_, 512);
  copy_cols_kernel<<<g1d(BB * TD_), 256, 0, stream>>>(GG, FUSCAT + TD_, BB, TD_, TD_, 512);
  launch_gemm(stream, FUSCAT, fus_W1, fus_b1, FUS1, BB, 1024, 512, 1024, 1);
  launch_gemm(stream, FUS1, fus_W2, fus_b2, FUSED, BB, TD_, 1024, TD_, 0);

  // -------- classifier --------
  launch_gemm(stream, FUSED, cls_W1, cls_b1, CLS1, BB, 128, TD_, 128, 0);
  ln_kernel<<<BB, 128, 0, stream>>>(CLS1, nullptr, cls_g, cls_b, CLS1, 128, 1); // LN + gelu
  launch_gemm(stream, CLS1, cls_W2, cls_b2, (float*)d_out, BB, 13, 128, 13, 0);
}
